// FidelityLossDetector_20005957665146
// MI455X (gfx1250) — matrix operand order repaired and checked by reading, not compiled, not run on hardware
//
#include <hip/hip_runtime.h>
#include <hip/hip_bf16.h>
#include <math.h>

typedef __attribute__((ext_vector_type(16))) _Float16 v16h;
typedef __attribute__((ext_vector_type(8)))  _Float16 v8h;
typedef __attribute__((ext_vector_type(8)))  float    v8f;
typedef __attribute__((ext_vector_type(4)))  unsigned int v4u;
typedef __attribute__((ext_vector_type(8)))  int      v8i;
typedef __attribute__((ext_vector_type(4)))  int      v4i;

#define B_N   131072
#define D_N   64
#define H_N   256
#define KP1   72    // w1t row stride in halves (64 + 8 pad)
#define KP2   264   // w2t/w3t row stride in halves (256 + 8 pad)
#define SP    264   // staging row stride in halves

#define W1T_BYTES (256 * KP1 * 2)   //  36864
#define W2T_BYTES (256 * KP2 * 2)   // 135168
#define W3T_BYTES (16  * KP2 * 2)   //   8448

#if !defined(__has_builtin)
#define __has_builtin(x) 0
#endif
#if !__has_builtin(__builtin_amdgcn_tensor_load_to_lds)
#define NO_TDM 1
#endif

// ---------------------------------------------------------------------------
// Kernel 0: zero the double accumulators
// ---------------------------------------------------------------------------
__global__ void fid_init_kernel(double* accum) {
    accum[0] = 0.0;
    accum[1] = 0.0;
}

// ---------------------------------------------------------------------------
// Kernel 1: per-row radii + double-precision sum / sumsq reduction
// ---------------------------------------------------------------------------
__global__ void fid_radii_kernel(const float* __restrict__ x,
                                 float* __restrict__ radii,
                                 double* __restrict__ accum) {
    int row = blockIdx.x * 256 + threadIdx.x;
    const float* xr = x + (size_t)row * D_N;
    float n2 = 0.f;
#pragma unroll
    for (int i = 0; i < D_N; i += 4) {
        float4 v = *(const float4*)(xr + i);
        n2 += v.x * v.x + v.y * v.y + v.z * v.z + v.w * v.w;
    }
    float den = fmaxf(1.f - n2, 1e-7f);
    float arg = fmaxf(1.f + 2.f * n2 / den, 1.f + 1e-7f);
    float r   = logf(arg + sqrtf(arg * arg - 1.f));   // arccosh
    radii[row] = r;

    __shared__ double sb[256];
    __shared__ double qb[256];
    sb[threadIdx.x] = (double)r;
    qb[threadIdx.x] = (double)r * (double)r;
    __syncthreads();
    for (int off = 128; off > 0; off >>= 1) {
        if (threadIdx.x < off) {
            sb[threadIdx.x] += sb[threadIdx.x + off];
            qb[threadIdx.x] += qb[threadIdx.x + off];
        }
        __syncthreads();
    }
    if (threadIdx.x == 0) {
        atomicAdd(&accum[0], sb[0]);
        atomicAdd(&accum[1], qb[0]);
    }
}

// ---------------------------------------------------------------------------
// Kernel 2: scalars: rich / metric / obj
// ---------------------------------------------------------------------------
__global__ void fid_scalars_kernel(const float* __restrict__ x,
                                   const double* __restrict__ accum,
                                   float* __restrict__ scalars) {
    if (threadIdx.x != 0 || blockIdx.x != 0) return;
    double s = accum[0], q = accum[1];
    double Bn = (double)B_N;
    double var = (q - s * s / Bn) / (Bn - 1.0);       // ddof = 1
    float varf = (float)var;
    scalars[0] = (varf < 0.003f) ? 0.3f : ((varf > 0.008f) ? 0.2f : 1.0f); // rich
    scalars[2] = 0.2f / (1.f + expf(-((varf - 0.1f) * 10.f)));             // obj

    float nm2[8];
    for (int i = 0; i < 8; i++) {
        float t = 0.f;
        for (int k = 0; k < D_N; k++) { float v = x[i * D_N + k]; t += v * v; }
        nm2[i] = t;
    }
    float hyp[28], euc[28];
    int p = 0;
    for (int i = 0; i < 8; i++) {
        for (int j = i + 1; j < 8; j++) {
            float d2 = 0.f;
            for (int k = 0; k < D_N; k++) {
                float d = x[i * D_N + k] - x[j * D_N + k];
                d2 += d * d;
            }
            float den = fmaxf((1.f - nm2[i]) * (1.f - nm2[j]), 1e-7f);
            float arg = fmaxf(1.f + 2.f * d2 / den, 1.f + 1e-7f);
            hyp[p] = logf(arg + sqrtf(arg * arg - 1.f));
            euc[p] = sqrtf(fmaxf(d2, 1e-7f));
            p++;
        }
    }
    float hm = 0.f, em = 0.f;
    for (p = 0; p < 28; p++) { hm += hyp[p]; em += euc[p]; }
    hm *= (1.f / 28.f); em *= (1.f / 28.f);
    float shh = 0.f, see = 0.f, she = 0.f;
    for (p = 0; p < 28; p++) {
        float hc = hyp[p] - hm, ec = euc[p] - em;
        shh += hc * hc; see += ec * ec; she += hc * ec;
    }
    float corr = she / (sqrtf(shh) * sqrtf(see) + 1e-8f);
    scalars[1] = (1.f - fabsf(corr)) * 0.4f;                               // metric
}

// ---------------------------------------------------------------------------
// Kernel 3: convert weights to f16 in the exact transposed+padded LDS image
//   ws layout:  w1h[256*72] | w2h[256*264] | w3h[16*264]
// ---------------------------------------------------------------------------
__global__ void fid_prep_kernel(const float* __restrict__ w1,
                                const float* __restrict__ w2,
                                const float* __restrict__ w3,
                                _Float16* __restrict__ w1h,
                                _Float16* __restrict__ w2h,
                                _Float16* __restrict__ w3h) {
    const int N1 = 256 * KP1;            // 18432
    const int N2 = 256 * KP2;            // 67584
    const int N3 = 16 * KP2;             //  4224
    for (int idx = blockIdx.x * 256 + threadIdx.x; idx < N1 + N2 + N3;
         idx += gridDim.x * 256) {
        if (idx < N1) {
            int n = idx / KP1, k = idx % KP1;
            w1h[idx] = (_Float16)((k < 64) ? w1[k * 256 + n] : 0.f);
        } else if (idx < N1 + N2) {
            int j = idx - N1;
            int n = j / KP2, k = j % KP2;
            w2h[j] = (_Float16)((k < 256) ? w2[k * 256 + n] : 0.f);
        } else {
            int j = idx - N1 - N2;
            int n = j / KP2, k = j % KP2;
            w3h[j] = (_Float16)((n < 5 && k < 256) ? w3[k * 5 + n] : 0.f);
        }
    }
}

// ---------------------------------------------------------------------------
// Tensor Data Mover: 1-D tile load, data_size = 8 bytes, n_units elements.
// D# per CDNA5 ISA.
// ---------------------------------------------------------------------------
#ifndef NO_TDM
__device__ inline void tdm_load_1d(unsigned int lds_addr, const void* gsrc,
                                   unsigned int n_units) {
    unsigned long long ga = (unsigned long long)(uintptr_t)gsrc;
    v4u g0;
    g0[0] = 1u;                                            // count=1, user D#
    g0[1] = lds_addr;                                      // lds_addr (bytes)
    g0[2] = (unsigned int)ga;                              // global_addr[31:0]
    g0[3] = (unsigned int)((ga >> 32) & 0x1FFFFFFu)        // global_addr[56:32]
          | (2u << 30);                                    // type = 2 (image)
    v8i g1;
    g1[0] = (int)(3u << 16);                               // data_size=3 (8B), no multicast
    g1[1] = (int)((n_units & 0xFFFFu) << 16);              // tensor_dim0[15:0]
    g1[2] = (int)((n_units >> 16) & 0xFFFFu);              // tensor_dim0[31:16]
    g1[3] = (int)((n_units & 0xFFFFu) << 16);              // tile_dim0 (16-bit)
    g1[4] = 0;                                             // tile_dim1/2 = 0
    g1[5] = (int)n_units;                                  // tensor_dim0_stride[31:0]
    g1[6] = 0;
    g1[7] = 0;
    v4i z4 = (v4i){0, 0, 0, 0};
#if __clang_major__ >= 23
    v8i z8 = (v8i){0, 0, 0, 0, 0, 0, 0, 0};
    __builtin_amdgcn_tensor_load_to_lds(g0, g1, z4, z4, z8, 0);
#else
    __builtin_amdgcn_tensor_load_to_lds(g0, g1, z4, z4, 0);
#endif
}
#endif

// ---------------------------------------------------------------------------
// WMMA helpers
// ---------------------------------------------------------------------------
__device__ inline v8f wmma16(v16h a, v16h b, v8f c) {
    return __builtin_amdgcn_wmma_f32_16x16x32_f16(
        false, a, false, b, (short)0, c, false, false);
}

__device__ inline v16h load16(const _Float16* p) {
    v8h lo = *(const v8h*)p;
    v8h hi = *(const v8h*)(p + 16);
    v16h r;
#pragma unroll
    for (int i = 0; i < 8; i++) { r[i] = lo[i]; r[i + 8] = hi[i]; }
    return r;
}

// A fragment (16x32 f16): half h of lane L holds element
//   (M = L%16, K = ks*32 + ((h&8)<<1) + ((L>=16)?8:0) + (h&7))
__device__ inline v16h load_afrag(const _Float16* st, int lane16, int hi, int ks) {
    const _Float16* ap = st + lane16 * SP + ks * 32 + (hi << 3);
    v8h c0 = *(const v8h*)ap;          // K = base .. base+7
    v8h c1 = *(const v8h*)(ap + 16);   // K = base+16 .. base+23
    v16h a;
#pragma unroll
    for (int i = 0; i < 8; i++) { a[i] = c0[i]; a[i + 8] = c1[i]; }
    return a;
}

__device__ inline void ln_silu_store(v8f* acc,
                                     const float* __restrict__ bias,
                                     const float* __restrict__ g,
                                     const float* __restrict__ be,
                                     _Float16* st, int lane16, int hi) {
#pragma unroll
    for (int t = 0; t < 16; t++) {
        float bv = bias[t * 16 + lane16];
#pragma unroll
        for (int r = 0; r < 8; r++) acc[t][r] += bv;
    }
    float mean[8], rstd[8];
#pragma unroll
    for (int r = 0; r < 8; r++) {
        float s = 0.f, q = 0.f;
#pragma unroll
        for (int t = 0; t < 16; t++) { float v = acc[t][r]; s += v; q += v * v; }
#pragma unroll
        for (int m = 1; m < 16; m <<= 1) {       // stays within each 16-lane half
            s += __shfl_xor(s, m, 32);
            q += __shfl_xor(q, m, 32);
        }
        float mu = s * (1.f / 256.f);
        mean[r] = mu;
        rstd[r] = rsqrtf(q * (1.f / 256.f) - mu * mu + 1e-5f);
    }
#pragma unroll
    for (int t = 0; t < 16; t++) {
        float gv = g[t * 16 + lane16];
        float ev = be[t * 16 + lane16];
#pragma unroll
        for (int r = 0; r < 8; r++) {
            float v = (acc[t][r] - mean[r]) * rstd[r] * gv + ev;
            v = v * (1.f / (1.f + __expf(-v)));   // SiLU
            st[(hi * 8 + r) * SP + t * 16 + lane16] = (_Float16)v;
        }
    }
}

// ---------------------------------------------------------------------------
// Kernel 4: fused MLP (WMMA) + final combine
//   8 waves / block, 16 rows / wave, 128 rows / block, 1024 blocks
// ---------------------------------------------------------------------------
__global__ __launch_bounds__(256)
void fid_mlp_kernel(const float* __restrict__ x,
                    const _Float16* __restrict__ w1h,
                    const _Float16* __restrict__ w2h,
                    const _Float16* __restrict__ w3h,
                    const float* __restrict__ b1, const float* __restrict__ g1,
                    const float* __restrict__ be1,
                    const float* __restrict__ b2, const float* __restrict__ g2,
                    const float* __restrict__ be2,
                    const float* __restrict__ b3,
                    const float* __restrict__ radii,
                    const float* __restrict__ scalars,
                    float* __restrict__ out) {
    // 242 KB static LDS (CDNA5: up to 320 KB per workgroup)
    __shared__ __attribute__((aligned(16))) _Float16 w1t[256 * KP1];
    __shared__ __attribute__((aligned(16))) _Float16 w2t[256 * KP2];
    __shared__ __attribute__((aligned(16))) _Float16 w3t[16 * KP2];
    __shared__ __attribute__((aligned(16))) _Float16 stg[8][16 * SP];

    const int tid = threadIdx.x;

#ifndef NO_TDM
    // Async-tensor fill of the three weight images (EXEC-independent, per-wave op)
    if (tid < 32) {
        tdm_load_1d((unsigned int)(uintptr_t)&w1t[0], w1h, W1T_BYTES / 8);
        tdm_load_1d((unsigned int)(uintptr_t)&w2t[0], w2h, W2T_BYTES / 8);
        tdm_load_1d((unsigned int)(uintptr_t)&w3t[0], w3h, W3T_BYTES / 8);
        __builtin_amdgcn_s_wait_tensorcnt(0);
    }
#else
    {
        const uint4* s1 = (const uint4*)w1h; uint4* d1 = (uint4*)w1t;
        for (int i = tid; i < W1T_BYTES / 16; i += 256) d1[i] = s1[i];
        const uint4* s2 = (const uint4*)w2h; uint4* d2 = (uint4*)w2t;
        for (int i = tid; i < W2T_BYTES / 16; i += 256) d2[i] = s2[i];
        const uint4* s3 = (const uint4*)w3h; uint4* d3 = (uint4*)w3t;
        for (int i = tid; i < W3T_BYTES / 16; i += 256) d3[i] = s3[i];
    }
#endif
    __syncthreads();

    const int wave   = tid >> 5;
    const int lane   = tid & 31;
    const int lane16 = lane & 15;
    const int hi     = lane >> 4;     // K-half selector for A/B fragments
    _Float16* st     = stg[wave];
    const int r0     = (blockIdx.x * 8 + wave) * 16;

    v8f acc[16];
#pragma unroll
    for (int t = 0; t < 16; t++) acc[t] = (v8f){0.f, 0.f, 0.f, 0.f, 0.f, 0.f, 0.f, 0.f};

    // ---- Layer 1: x(16x64) @ w1(64x256) ----
    {
        const float* xr = x + (size_t)(r0 + lane16) * D_N;
        v16h a0, a1;
#pragma unroll
        for (int h = 0; h < 16; h++) {
            int kk = ((h & 8) << 1) + (hi << 3) + (h & 7);
            a0[h] = (_Float16)xr[kk];
            a1[h] = (_Float16)xr[32 + kk];
        }
#pragma unroll
        for (int t = 0; t < 16; t++) {
            const _Float16* bp = w1t + (t * 16 + lane16) * KP1 + (hi << 3);
            acc[t] = wmma16(a0, load16(bp),      acc[t]);   // K = 0..31
            acc[t] = wmma16(a1, load16(bp + 32), acc[t]);   // K = 32..63
        }
    }
    ln_silu_store(acc, b1, g1, be1, st, lane16, hi);

    // ---- Layer 2: h1(16x256) @ w2(256x256) ----
#pragma unroll
    for (int t = 0; t < 16; t++) acc[t] = (v8f){0.f, 0.f, 0.f, 0.f, 0.f, 0.f, 0.f, 0.f};
#pragma unroll
    for (int ks = 0; ks < 8; ks++) {
        v16h a = load_afrag(st, lane16, hi, ks);
#pragma unroll
        for (int t = 0; t < 16; t++) {
            const _Float16* bp = w2t + (t * 16 + lane16) * KP2 + ks * 32 + (hi << 3);
            acc[t] = wmma16(a, load16(bp), acc[t]);
        }
    }
    ln_silu_store(acc, b2, g2, be2, st, lane16, hi);

    // ---- Layer 3: h2(16x256) @ w3(256x16-padded) ----
    v8f acc3 = (v8f){0.f, 0.f, 0.f, 0.f, 0.f, 0.f, 0.f, 0.f};
#pragma unroll
    for (int ks = 0; ks < 8; ks++) {
        v16h a = load_afrag(st, lane16, hi, ks);
        const _Float16* bp = w3t + lane16 * KP2 + ks * 32 + (hi << 3);
        acc3 = wmma16(a, load16(bp), acc3);
    }

    // sigmoid scores -> 16x16 float grid in staging for per-row combine
    float* sg = (float*)st;
    {
        float b3v = (lane16 < 5) ? b3[lane16] : 0.f;
#pragma unroll
        for (int r = 0; r < 8; r++) {
            float v = acc3[r] + b3v;
            sg[(hi * 8 + r) * 16 + lane16] = 1.f / (1.f + __expf(-v));
        }
    }

    // ---- final combine: lanes 0..15 each own one row ----
    if (lane < 16) {
        int row = r0 + lane;
        float s0 = sg[lane * 16 + 0];
        float s1 = sg[lane * 16 + 1];
        float s2 = sg[lane * 16 + 2];
        float s3 = sg[lane * 16 + 3];
        float s4 = sg[lane * 16 + 4];
        float total = (s0 + s1 + s2 + s3 + s4) * 0.2f;

        float r = radii[row];
        float recon = (r * 10.f <= 2.f) ? 0.5f : 0.f;  // 9-clip(r*10,0,9) >= 7
        float kl = (__expf(-r * 5.f) +
                    1.f / (1.f + __expf(-((r - 0.95f) * 20.f)))) * 0.3f;
        float rich = scalars[0], metric = scalars[1], obj = scalars[2];

        out[0 * B_N + row] = s0 + recon;
        out[1 * B_N + row] = s1 + rich;
        out[2 * B_N + row] = s2 + metric;
        out[3 * B_N + row] = s3 + kl;
        out[4 * B_N + row] = s4 + obj;
        out[5 * B_N + row] = total;
    }
}

// ---------------------------------------------------------------------------
// launch
// ---------------------------------------------------------------------------
extern "C" void kernel_launch(void* const* d_in, const int* in_sizes, int n_in,
                              void* d_out, int out_size, void* d_ws, size_t ws_size,
                              hipStream_t stream) {
    const float* x   = (const float*)d_in[0];
    const float* w1  = (const float*)d_in[1];
    const float* b1  = (const float*)d_in[2];
    const float* g1  = (const float*)d_in[3];
    const float* be1 = (const float*)d_in[4];
    const float* w2  = (const float*)d_in[5];
    const float* b2  = (const float*)d_in[6];
    const float* g2  = (const float*)d_in[7];
    const float* be2 = (const float*)d_in[8];
    const float* w3  = (const float*)d_in[9];
    const float* b3  = (const float*)d_in[10];
    float* out = (float*)d_out;

    char* ws = (char*)d_ws;
    double* accum   = (double*)ws;                     // 16 B: sum, sumsq
    float*  scalars = (float*)(ws + 16);               // rich, metric, obj
    float*  radii   = (float*)(ws + 64);               // B floats (524288 B)
    _Float16* w1h   = (_Float16*)(ws + 524352);        // 36864 B
    _Float16* w2h   = (_Float16*)(ws + 524352 + W1T_BYTES);              // 135168 B
    _Float16* w3h   = (_Float16*)(ws + 524352 + W1T_BYTES + W2T_BYTES);  // 8448 B

    fid_init_kernel<<<1, 1, 0, stream>>>(accum);
    fid_radii_kernel<<<B_N / 256, 256, 0, stream>>>(x, radii, accum);
    fid_scalars_kernel<<<1, 64, 0, stream>>>(x, accum, scalars);
    fid_prep_kernel<<<353, 256, 0, stream>>>(w1, w2, w3, w1h, w2h, w3h);
    fid_mlp_kernel<<<B_N / 128, 256, 0, stream>>>(
        x, w1h, w2h, w3h, b1, g1, be1, b2, g2, be2, b3, radii, scalars, out);
}
